// LiquidDecoderLayer_74947179315327
// MI455X (gfx1250) — matrix operand order repaired and checked by reading, not compiled, not run on hardware
//
#include <hip/hip_runtime.h>

// ---------------------------------------------------------------------------
// LiquidDecoderLayer for MI455X (gfx1250, wave32, WMMA f32_16x16x32_f16,
// async global->LDS staging, persistent-workgroup CfC recurrence)
// ---------------------------------------------------------------------------

typedef _Float16 h16;
typedef __attribute__((ext_vector_type(16))) _Float16 v16h;
typedef __attribute__((ext_vector_type(8)))  float    v8f;

#define WMMA_F16(a, b, c) \
  __builtin_amdgcn_wmma_f32_16x16x32_f16(false, (a), false, (b), (short)0, (c), false, false)

// ---- dims ----
#define BD   8
#define LD   1024
#define DD   512
#define LMEM 512
#define DFF  2048
#define ROWS (BD * LD)     // 8192
#define KVROWS (BD * LMEM) // 4096

// ---------------------------------------------------------------------------
// WMMA fragment helpers (A/B: contiguous k = 16h + s, not the ISA table order; C/D per ISA, wave32)
// ---------------------------------------------------------------------------

// A: 16x32 f16 from row-major f16 buffer (LDS): lane->M, K pairs per VGPR,
// lanes 0-15 hold K 0-15, lanes 16-31 hold K 16-31 (contiguous).
__device__ __forceinline__ v16h load_a_frag(const h16* src, int ld) {
  int lane = threadIdx.x & 31;
  int m = lane & 15;
  int hh = lane >> 4;
  v16h a;
#pragma unroll
  for (int r = 0; r < 8; ++r) {
    int k0 = 16 * hh + 2 * r;
    a[2 * r]     = src[m * ld + k0];
    a[2 * r + 1] = src[m * ld + k0 + 1];
  }
  return a;
}

// A fragment from a row-major f32 buffer (LDS panel staged by async DMA).
__device__ __forceinline__ v16h load_a_frag_f32(const float* src, int ld) {
  int lane = threadIdx.x & 31;
  int m = lane & 15;
  int hh = lane >> 4;
  v16h a;
#pragma unroll
  for (int r = 0; r < 8; ++r) {
    int k0 = 16 * hh + 2 * r;
    a[2 * r]     = (h16)src[m * ld + k0];
    a[2 * r + 1] = (h16)src[m * ld + k0 + 1];
  }
  return a;
}

// B: 32x16 f16 where B[k][n] = base[off + n*ld + k] (w is [N][K] -> X@W^T).
// base is a direct kernel-arg pointer + 32-bit offset -> global_load saddr form.
__device__ __forceinline__ v16h load_b_frag_f16(const h16* __restrict__ base,
                                                int off, int ld) {
  int lane = threadIdx.x & 31;
  int n = lane & 15;
  int hh = lane >> 4;
  const h16* p = base + off + n * ld + 16 * hh;
  v16h b;
#pragma unroll
  for (int e = 0; e < 16; ++e) b[e] = p[e];
  return b;
}

// Same B layout, f32 source converted on the fly: B[k][n] = w[n*ld + k].
__device__ __forceinline__ v16h load_b_frag_f32(const float* w, int ld) {
  int lane = threadIdx.x & 31;
  int n = lane & 15;
  int hh = lane >> 4;
  const float* p = w + n * ld + 16 * hh;
  v16h b;
#pragma unroll
  for (int e = 0; e < 16; ++e) b[e] = (h16)p[e];
  return b;
}

// B[k][n] = w[k*ld + n]  (row-major K x N f32 source, e.g. V matrix).
__device__ __forceinline__ v16h load_b_frag_f32_rm(const float* w, int ld) {
  int lane = threadIdx.x & 31;
  int n = lane & 15;
  int hh = lane >> 4;
  v16h b;
#pragma unroll
  for (int e = 0; e < 16; ++e) b[e] = (h16)w[(16 * hh + e) * ld + n];
  return b;
}

// C/D: 16x16 f32: lane -> N, vgpr r -> M=r (lanes 0-15) / M=r+8 (lanes 16-31).
__device__ __forceinline__ void store_c_frag(float* dst, int ld, v8f c, int mlim) {
  int lane = threadIdx.x & 31;
  int n = lane & 15;
  int hh = lane >> 4;
#pragma unroll
  for (int r = 0; r < 8; ++r) {
    int m = r + 8 * hh;
    if (m < mlim) dst[m * ld + n] = c[r];
  }
}

// ---------------------------------------------------------------------------
// Weight prep
// ---------------------------------------------------------------------------

__global__ void cvt_f32_to_f16(const float* __restrict__ s, h16* __restrict__ d, int n) {
  int i = blockIdx.x * blockDim.x + threadIdx.x;
  if (i < n) d[i] = (h16)s[i];
}

// Concatenated CfC layer weight [3h rows padded][cat padded] f16:
// rows [0,h): ff1*mask ; [h,2h): ff2*mask ; [2h,3h): ta+tb (sigmoid-gate fold).
__global__ void prep_cfc_w(const float* __restrict__ f1, const float* __restrict__ f2,
                           const float* __restrict__ ta, const float* __restrict__ tb,
                           const float* __restrict__ mk, h16* __restrict__ wc,
                           int h, int cat, int catp, int rows3p) {
  int i = blockIdx.x * blockDim.x + threadIdx.x;
  if (i >= rows3p * catp) return;
  int r = i / catp, k = i - r * catp;
  float v = 0.f;
  if (k < cat && r < 3 * h) {
    if (r < h)            v = f1[r * cat + k] * mk[r * cat + k];
    else if (r < 2 * h)   v = f2[(r - h) * cat + k] * mk[(r - h) * cat + k];
    else                  v = ta[(r - 2 * h) * cat + k] + tb[(r - 2 * h) * cat + k];
  }
  wc[i] = (h16)v;
}

__global__ void prep_cfc_b(const float* __restrict__ b1, const float* __restrict__ b2,
                           const float* __restrict__ ba, const float* __restrict__ bb,
                           float* __restrict__ bc, int h, int rows3p) {
  int r = blockIdx.x * blockDim.x + threadIdx.x;
  if (r >= rows3p) return;
  float v = 0.f;
  if (r < h)          v = b1[r];
  else if (r < 2 * h) v = b2[r - h];
  else if (r < 3 * h) v = ba[r - 2 * h] + bb[r - 2 * h];
  bc[r] = v;
}

// ---------------------------------------------------------------------------
// LayerNorm, width fixed 512. 8 waves/block, one wave per row.
// ---------------------------------------------------------------------------

__global__ void __launch_bounds__(256) layernorm512(
    const float* __restrict__ x, const float* __restrict__ w,
    const float* __restrict__ b, float* __restrict__ y, int rows) {
  int wid = threadIdx.x >> 5;
  int lane = threadIdx.x & 31;
  int row = blockIdx.x * 8 + wid;
  if (row >= rows) return;
  const float* xr = x + (size_t)row * DD;
  float v[16];
  float s = 0.f;
#pragma unroll
  for (int i = 0; i < 16; ++i) { v[i] = xr[lane * 16 + i]; s += v[i]; }
#pragma unroll
  for (int o = 16; o >= 1; o >>= 1) s += __shfl_xor(s, o, 32);
  float m = s * (1.f / DD);
  float vs = 0.f;
#pragma unroll
  for (int i = 0; i < 16; ++i) { float d = v[i] - m; vs += d * d; }
#pragma unroll
  for (int o = 16; o >= 1; o >>= 1) vs += __shfl_xor(vs, o, 32);
  float rs = rsqrtf(vs * (1.f / DD) + 1e-5f);
  float* yr = y + (size_t)row * DD;
#pragma unroll
  for (int i = 0; i < 16; ++i) {
    int c = lane * 16 + i;
    yr[c] = (v[i] - m) * rs * w[c] + b[c];
  }
}

// ---------------------------------------------------------------------------
// CfC layer body. K-chunk loop is OUTER (unroll 1: avoids the unroll-and-
// spill of A fragments seen when K was fully unrolled); each wave keeps up to
// ceil(NTILES/32) accumulators resident and reuses one A fragment per K-chunk
// across all its tiles (A pipelined one chunk ahead). Weight pointer is a
// direct kernel arg -> global_load_b128 saddr form.
// ---------------------------------------------------------------------------

template <int H, int INF, int CATN, int CATP, int NTILES, int SPL>
__device__ __forceinline__ void cfc_layer(
    const h16* __restrict__ wcl, const float* __restrict__ bcl,
    float* hstate, float* inbuf, float* ybuf, h16* xc) {
  const int tid = threadIdx.x;
  const int wave = tid >> 5;

  // build xc (f16) = [input, old h-slice], zero-padded to 16 rows / CATP cols
  for (int i = tid; i < 16 * CATP; i += 1024) {
    int bb = i / CATP, k = i - bb * CATP;
    float v = 0.f;
    if (bb < 8) {
      if (k < INF)       v = inbuf[bb * 512 + k];
      else if (k < CATN) v = hstate[bb * 1024 + SPL + (k - INF)];
    }
    xc[i] = (h16)v;
  }
  __syncthreads();

  constexpr int NKC = CATP / 32;
  constexpr int TPW = (NTILES + 31) / 32;  // tiles per wave (max 3)

  v8f acc[TPW];
#pragma unroll
  for (int j = 0; j < TPW; ++j) acc[j] = {};
  if (wave < NTILES)  // wave-uniform; idle waves skip straight to barrier
  {
    __builtin_prefetch(wcl + (size_t)wave * 16 * CATP, 0, 1);
    v16h a = load_a_frag(xc, CATP);
#pragma unroll 1
    for (int kc = 0; kc < NKC; ++kc) {
      v16h an = a;
      if (kc + 1 < NKC) an = load_a_frag(xc + (kc + 1) * 32, CATP);
#pragma unroll
      for (int j = 0; j < TPW; ++j) {
        int nt = wave + 32 * j;
        if (nt < NTILES) {  // wave-uniform
          v16h bf = load_b_frag_f16(wcl, nt * 16 * CATP + kc * 32, CATP);
          acc[j] = WMMA_F16(a, bf, acc[j]);
        }
      }
      a = an;
    }
    int lane = tid & 31, n = lane & 15, hh = lane >> 4;
#pragma unroll
    for (int j = 0; j < TPW; ++j) {
      int nt = wave + 32 * j;
      if (nt < NTILES) {
        float bias = bcl[nt * 16 + n];
#pragma unroll
        for (int r = 0; r < 8; ++r) {
          int m = r + 8 * hh;
          if (m < 8) ybuf[m * 1536 + nt * 16 + n] = acc[j][r] + bias;
        }
      }
    }
  }
  __syncthreads();

  // gating: h_new = tanh(f1)*(1-sig) + sig*tanh(f2)
  for (int i = tid; i < 8 * H; i += 1024) {
    int bb = i / H, n = i - bb * H;
    float f1 = tanhf(ybuf[bb * 1536 + n]);
    float f2 = tanhf(ybuf[bb * 1536 + H + n]);
    float tt = 1.f / (1.f + __expf(-ybuf[bb * 1536 + 2 * H + n]));
    float hv = f1 * (1.f - tt) + tt * f2;
    hstate[bb * 1024 + SPL + n] = hv;
    inbuf[bb * 512 + n] = hv;  // becomes next layer's input
  }
  __syncthreads();
}

// Persistent CfC scan: ONE workgroup (32 waves), h-state in LDS, weights
// streamed from L2 each step. Sequential over t (true recurrence).
__global__ void __launch_bounds__(1024) cfc_scan(
    const float* __restrict__ xn, const float* __restrict__ x,
    const h16* __restrict__ wc0, const h16* __restrict__ wc1, const h16* __restrict__ wc2,
    const float* __restrict__ bc0, const float* __restrict__ bc1, const float* __restrict__ bc2,
    float* __restrict__ x1, float* __restrict__ hn_out) {
  extern __shared__ char smem_raw[];
  float* hstate = (float*)smem_raw;              // 8 x 1024 f32   (32 KB)
  float* inbuf  = hstate + 8 * 1024;             // 8 x 512  f32   (16 KB)
  float* ybuf   = inbuf + 8 * 512;               // 8 x 1536 f32   (48 KB)
  h16*   xc     = (h16*)(ybuf + 8 * 1536);       // 16 x 832 f16   (26 KB)

  const int tid = threadIdx.x;

  for (int i = tid; i < 8 * 1024; i += 1024) hstate[i] = 0.f;
  __syncthreads();

  for (int t = 0; t < LD; ++t) {
    for (int i = tid; i < 8 * 512; i += 1024) {
      int bb = i >> 9, c = i & 511;
      inbuf[i] = xn[((size_t)bb * LD + t) * DD + c];
    }
    __syncthreads();

    cfc_layer<308, 512, 820, 832, 58, 0  >(wc0, bc0, hstate, inbuf, ybuf, xc);
    cfc_layer<204, 308, 512, 512, 39, 308>(wc1, bc1, hstate, inbuf, ybuf, xc);
    cfc_layer<512, 204, 716, 736, 96, 512>(wc2, bc2, hstate, inbuf, ybuf, xc);

    // motor output (layer 2, width 512) + residual -> x1
    for (int i = tid; i < 8 * 512; i += 1024) {
      int bb = i >> 9, c = i & 511;
      size_t gi = ((size_t)bb * LD + t) * DD + c;
      x1[gi] = x[gi] + inbuf[i];
    }
    __syncthreads();
  }

  for (int i = tid; i < 8 * 1024; i += 1024) hn_out[i] = hstate[i];
}

// ---------------------------------------------------------------------------
// Generic GEMM: Y[M][N] = act(X[M][K] @ W^T + bias) (+ res). W is f16 [N][K].
// Block = 256 thr (8 waves) -> 16(M) x 128(N) tile.
// A panel (16 x 128 f32) staged via GLOBAL_LOAD_ASYNC_TO_LDS_B128 with
// double buffering: DMA for chunk kc+1 overlaps WMMA on chunk kc; chunk
// readiness enforced with s_wait_asynccnt (async ops complete in order).
// ---------------------------------------------------------------------------

#define KSTEP 128

__global__ void __launch_bounds__(256) gemm_xwT(
    const float* __restrict__ X, const h16* __restrict__ W,
    const float* __restrict__ bias, const float* __restrict__ res,
    float* __restrict__ Y, int M, int N, int K, int act) {
  extern __shared__ char smem_raw[];
  float* at = (float*)smem_raw;  // 2 x (16 x 128) f32 = 16 KB
  int m0 = blockIdx.y * 16;
  int wave = threadIdx.x >> 5;
  int ntile = blockIdx.x * 128 + wave * 16;

  // issue 16B-granule async copies of one 16x128 f32 panel (512 granules,
  // 2 per thread). ASYNCcnt += 2 per wave per panel.
  auto issue_panel = [&](int bufi, int kc) {
    for (int i = threadIdx.x; i < 512; i += 256) {
      int row = i >> 5;        // 0..15
      int c16 = i & 31;        // 16-byte granule within the 512B row
      const float* g = X + (size_t)(m0 + row) * K + kc + c16 * 4;
      float* l = at + bufi * (16 * KSTEP) + row * KSTEP + c16 * 4;
      unsigned lds_off = (unsigned)(size_t)(void*)l;  // LDS aperture low bits
      asm volatile("global_load_async_to_lds_b128 %0, %1, off"
                   :: "v"(lds_off), "v"(g)
                   : "memory");
    }
  };

  v8f acc = {};
  issue_panel(0, 0);
  int bufi = 0;
  for (int kc = 0; kc < K; kc += KSTEP, bufi ^= 1) {
    bool more = (kc + KSTEP < K);
    if (more) {
      issue_panel(bufi ^ 1, kc + KSTEP);
      asm volatile("s_wait_asynccnt 2" ::: "memory");  // panel kc resident
    } else {
      asm volatile("s_wait_asynccnt 0" ::: "memory");
    }
    __syncthreads();
    const float* ap = at + bufi * (16 * KSTEP);
    const int wbase = ntile * K + kc;
#pragma unroll
    for (int s = 0; s < KSTEP / 32; ++s) {
      v16h a = load_a_frag_f32(ap + s * 32, KSTEP);
      v16h bf = load_b_frag_f16(W, wbase + s * 32, K);
      acc = WMMA_F16(a, bf, acc);
    }
    __syncthreads();  // all waves done with this panel before DMA reuses it
  }

  int lane = threadIdx.x & 31, n = lane & 15, hh = lane >> 4;
  int ng = ntile + n;
  float bv = bias ? bias[ng] : 0.f;
#pragma unroll
  for (int r = 0; r < 8; ++r) {
    int m = m0 + r + 8 * hh;
    float v = acc[r] + bv;
    if (act == 1) v = 0.5f * v * (1.f + erff(v * 0.70710678118654752f)); // exact GELU
    if (res) v += res[(size_t)m * N + ng];
    Y[(size_t)m * N + ng] = v;
  }
}

// ---------------------------------------------------------------------------
// Cross-attention: one block per (qtile16, head, batch). Full-row softmax
// (16 x 512 scores in LDS), QK^T and P@V via WMMA.
// ---------------------------------------------------------------------------

__global__ void __launch_bounds__(256) attn_kernel(
    const float* __restrict__ q, const float* __restrict__ k,
    const float* __restrict__ v, float* __restrict__ og) {
  extern __shared__ char smem_raw[];
  h16*   qt   = (h16*)smem_raw;                       // 16 x 64 f16
  float* sbuf = (float*)(smem_raw + 16 * 64 * 2);     // 16 x 512 f32
  h16*   pbuf = (h16*)(sbuf + 16 * 512);              // 16 x 512 f16

  int b = blockIdx.z, head = blockIdx.y, q0 = blockIdx.x * 16;
  int wave = threadIdx.x >> 5;
  const float scale = 0.125f;   // 1/sqrt(64)

  for (int i = threadIdx.x; i < 16 * 64; i += 256) {
    int r = i >> 6, c = i & 63;
    qt[i] = (h16)(q[((size_t)b * LD + q0 + r) * DD + head * 64 + c] * scale);
  }
  __syncthreads();

  // scores: 32 key tiles of 16
  for (int nt = wave; nt < 32; nt += 8) {
    v8f acc = {};
    const float* kbase = k + ((size_t)b * LMEM + nt * 16) * DD + head * 64;
#pragma unroll
    for (int kc = 0; kc < 2; ++kc) {
      v16h a = load_a_frag(qt + kc * 32, 64);
      v16h bf = load_b_frag_f32(kbase + kc * 32, DD);
      acc = WMMA_F16(a, bf, acc);
    }
    store_c_frag(sbuf + nt * 16, 512, acc, 16);
  }
  __syncthreads();

  // softmax over 512 keys; 16 threads per row (xor-shuffle stays in half-wave)
  {
    int row = threadIdx.x >> 4;
    int sub = threadIdx.x & 15;
    float* srow = sbuf + row * 512;
    float mx = -1e30f;
    for (int c = sub; c < 512; c += 16) mx = fmaxf(mx, srow[c]);
#pragma unroll
    for (int o = 8; o >= 1; o >>= 1) mx = fmaxf(mx, __shfl_xor(mx, o, 32));
    float sm = 0.f;
    for (int c = sub; c < 512; c += 16) {
      float e = __expf(srow[c] - mx);
      srow[c] = e;
      sm += e;
    }
#pragma unroll
    for (int o = 8; o >= 1; o >>= 1) sm += __shfl_xor(sm, o, 32);
    float inv = 1.f / sm;
    for (int c = sub; c < 512; c += 16) pbuf[row * 512 + c] = (h16)(srow[c] * inv);
  }
  __syncthreads();

  // O = P @ V : 4 output tiles of 16 over dh=64 (waves 0-3, wave-uniform)
  if (wave < 4) {
    v8f acc = {};
    const float* vbase = v + (size_t)b * LMEM * DD + head * 64 + wave * 16;
    for (int kc = 0; kc < 16; ++kc) {
      v16h a = load_a_frag(pbuf + kc * 32, 512);
      v16h bf = load_b_frag_f32_rm(vbase + (size_t)kc * 32 * DD, DD);
      acc = WMMA_F16(a, bf, acc);
    }
    int lane = threadIdx.x & 31, n = lane & 15, hh = lane >> 4;
#pragma unroll
    for (int r = 0; r < 8; ++r) {
      int m = r + 8 * hh;
      og[((size_t)b * LD + q0 + m) * DD + head * 64 + wave * 16 + n] = acc[r];
    }
  }
}

// ---------------------------------------------------------------------------
// Host-side orchestration
// ---------------------------------------------------------------------------

extern "C" void kernel_launch(void* const* d_in, const int* in_sizes, int n_in,
                              void* d_out, int out_size, void* d_ws, size_t ws_size,
                              hipStream_t stream) {
  (void)in_sizes; (void)n_in; (void)out_size; (void)ws_size;

  // ---- inputs (setup_inputs dict order, lists flattened in sequence) ----
  const float* x       = (const float*)d_in[0];
  const float* memory  = (const float*)d_in[1];
  const float* cfc_nw  = (const float*)d_in[2];
  const float* cfc_nb  = (const float*)d_in[3];
  const float* ff1w[3] = {(const float*)d_in[4],  (const float*)d_in[5],  (const float*)d_in[6]};
  const float* ff1b[3] = {(const float*)d_in[7],  (const float*)d_in[8],  (const float*)d_in[9]};
  const float* ff2w[3] = {(const float*)d_in[10], (const float*)d_in[11], (const float*)d_in[12]};
  const float* ff2b[3] = {(const float*)d_in[13], (const float*)d_in[14], (const float*)d_in[15]};
  const float* taw[3]  = {(const float*)d_in[16], (const float*)d_in[17], (const float*)d_in[18]};
  const float* tab[3]  = {(const float*)d_in[19], (const float*)d_in[20], (const float*)d_in[21]};
  const float* tbw[3]  = {(const float*)d_in[22], (const float*)d_in[23], (const float*)d_in[24]};
  const float* tbb[3]  = {(const float*)d_in[25], (const float*)d_in[26], (const float*)d_in[27]};
  const float* mask[3] = {(const float*)d_in[28], (const float*)d_in[29], (const float*)d_in[30]};
  const float* attn_nw = (const float*)d_in[31];
  const float* attn_nb = (const float*)d_in[32];
  const float* ipw     = (const float*)d_in[33];
  const float* ipb     = (const float*)d_in[34];
  const float* opw     = (const float*)d_in[35];
  const float* opb     = (const float*)d_in[36];
  const float* nw      = (const float*)d_in[37];
  const float* nb      = (const float*)d_in[38];
  const float* w1      = (const float*)d_in[39];
  const float* b1      = (const float*)d_in[40];
  const float* w2      = (const float*)d_in[41];
  const float* b2      = (const float*)d_in[42];

  float* outf = (float*)d_out;                 // x2 then final output
  float* hn_out = outf + (size_t)BD * LD * DD; // hn tail of tuple output

  // ---- workspace carve ----
  char* ws = (char*)d_ws;
  size_t off = 0;
  auto alloc = [&](size_t bytes) -> void* {
    off = (off + 255) & ~(size_t)255;
    void* p = ws + off;
    off += bytes;
    return p;
  };
  const int Hh[3]   = {308, 204, 512};
  const int CAT[3]  = {820, 512, 716};
  const int CATP[3] = {832, 512, 736};
  const int R3P[3]  = {928, 624, 1536};

  h16* wc[3]; float* bc[3];
  for (int l = 0; l < 3; ++l) {
    wc[l] = (h16*)alloc((size_t)R3P[l] * CATP[l] * sizeof(h16));
    bc[l] = (float*)alloc((size_t)R3P[l] * sizeof(float));
  }
  h16* wip = (h16*)alloc((size_t)3 * DD * DD * sizeof(h16));   // q|k|v
  h16* wo  = (h16*)alloc((size_t)DD * DD * sizeof(h16));
  h16* w1h = (h16*)alloc((size_t)DFF * DD * sizeof(h16));
  h16* w2h = (h16*)alloc((size_t)DD * DFF * sizeof(h16));
  float* xn    = (float*)alloc((size_t)ROWS * DD * 4);   // reused as xn2
  float* x1b   = (float*)alloc((size_t)ROWS * DD * 4);
  float* hq    = (float*)alloc((size_t)ROWS * DD * 4);
  float* qb    = (float*)alloc((size_t)ROWS * DD * 4);
  float* kb    = (float*)alloc((size_t)KVROWS * DD * 4);
  float* vb    = (float*)alloc((size_t)KVROWS * DD * 4);
  float* obuf  = (float*)alloc((size_t)ROWS * DD * 4);
  float* hid   = (float*)alloc((size_t)ROWS * DFF * 4);

  // ---- weight prep ----
  for (int l = 0; l < 3; ++l) {
    int tot = R3P[l] * CATP[l];
    prep_cfc_w<<<(tot + 255) / 256, 256, 0, stream>>>(
        ff1w[l], ff2w[l], taw[l], tbw[l], mask[l], wc[l], Hh[l], CAT[l], CATP[l], R3P[l]);
    prep_cfc_b<<<(R3P[l] + 255) / 256, 256, 0, stream>>>(
        ff1b[l], ff2b[l], tab[l], tbb[l], bc[l], Hh[l], R3P[l]);
  }
  cvt_f32_to_f16<<<(3 * DD * DD + 255) / 256, 256, 0, stream>>>(ipw, wip, 3 * DD * DD);
  cvt_f32_to_f16<<<(DD * DD + 255) / 256, 256, 0, stream>>>(opw, wo, DD * DD);
  cvt_f32_to_f16<<<(DFF * DD + 255) / 256, 256, 0, stream>>>(w1, w1h, DFF * DD);
  cvt_f32_to_f16<<<(DD * DFF + 255) / 256, 256, 0, stream>>>(w2, w2h, DD * DFF);

  // ---- CfC block ----
  layernorm512<<<ROWS / 8, 256, 0, stream>>>(x, cfc_nw, cfc_nb, xn, ROWS);
  size_t scan_smem = 8 * 1024 * 4 + 8 * 512 * 4 + 8 * 1536 * 4 + 16 * 832 * 2; // 124928 B
  cfc_scan<<<1, 1024, scan_smem, stream>>>(xn, x, wc[0], wc[1], wc[2],
                                           bc[0], bc[1], bc[2], x1b, hn_out);

  // ---- cross attention ----
  layernorm512<<<ROWS / 8, 256, 0, stream>>>(x1b, attn_nw, attn_nb, hq, ROWS);
  size_t gemm_smem = 2 * 16 * KSTEP * sizeof(float);  // 16 KB
  gemm_xwT<<<dim3(DD / 128, ROWS / 16), 256, gemm_smem, stream>>>(
      hq, wip, ipb, nullptr, qb, ROWS, DD, DD, 0);
  gemm_xwT<<<dim3(DD / 128, KVROWS / 16), 256, gemm_smem, stream>>>(
      memory, wip + (size_t)DD * DD, ipb + DD, nullptr, kb, KVROWS, DD, DD, 0);
  gemm_xwT<<<dim3(DD / 128, KVROWS / 16), 256, gemm_smem, stream>>>(
      memory, wip + (size_t)2 * DD * DD, ipb + 2 * DD, nullptr, vb, KVROWS, DD, DD, 0);
  size_t attn_smem = 16 * 64 * 2 + 16 * 512 * 4 + 16 * 512 * 2;  // 51200 B
  attn_kernel<<<dim3(LD / 16, 8, BD), 256, attn_smem, stream>>>(qb, kb, vb, obuf);
  gemm_xwT<<<dim3(DD / 128, ROWS / 16), 256, gemm_smem, stream>>>(
      obuf, wo, opb, x1b, outf, ROWS, DD, DD, 0);   // x2 = x1 + o@Wo^T + b

  // ---- FFN ----
  layernorm512<<<ROWS / 8, 256, 0, stream>>>(outf, nw, nb, xn, ROWS);
  gemm_xwT<<<dim3(DFF / 128, ROWS / 16), 256, gemm_smem, stream>>>(
      xn, w1h, b1, nullptr, hid, ROWS, DFF, DD, 1);              // GELU
  gemm_xwT<<<dim3(DD / 128, ROWS / 16), 256, gemm_smem, stream>>>(
      hid, w2h, b2, outf, outf, ROWS, DD, DFF, 0);               // out = x2 + ffn
}
